// OCRYOLOv8Loss_90984587199022
// MI455X (gfx1250) — matrix operand order repaired and checked by reading, not compiled, not run on hardware
//
#include <hip/hip_runtime.h>
#include <hip/hip_fp16.h>

// ---------------------------------------------------------------------------
// OCR-YOLOv8 loss on MI455X (gfx1250, wave32).
// K1 (decode) is HBM-bound (~70 MB of boxes @ 23.3 TB/s); the softmax
// expectation is mapped onto v_wmma_f32_16x16x32_f16 with probabilities
// normalized pre-WMMA (one extra shfl_xor(16) + v_rcp per side), so D's
// columns 0/1 are the expected values directly — no post-WMMA shuffles or
// IEEE divides. K2 (CIoU) is L2-resident and VALU-bound; fast v_rcp_f32
// division replaces the div_scale/div_fixup chains.
// ---------------------------------------------------------------------------

typedef __attribute__((ext_vector_type(16))) _Float16 v16h;
typedef __attribute__((ext_vector_type(8)))  float    v8f;

#define REGMAX 16
#define TOPK   10
#define EPS_   1e-7f
#define NEG_INF_ (-3.402823466e38f)

__device__ __forceinline__ float fast_rcp(float x) {
  return __builtin_amdgcn_rcpf(x);               // v_rcp_f32 (~1 ulp)
}
__device__ __forceinline__ float fast_div(float a, float b) {
  return a * __builtin_amdgcn_rcpf(b);
}

// ---------------------------------------------------------------- zero fill
__global__ void zero_f32(float* __restrict__ p, int n) {
  int i = blockIdx.x * blockDim.x + threadIdx.x;
  if (i < n) p[i] = 0.0f;
}

// ---------------------------------------------------------------- K1: decode
// boxes: [B, 4*16, N] fp32.  pb out: [B, N, 4] fp32 (xyxy expected values).
// One wave = 16 anchors, two WMMAs (sides 0/1 and 2/3).
// A-matrix 16-bit layout: lanes 0-15 hold M=lane, K=0..7 (V0-3) & 16..23
// (V4-7); lanes 16-31 hold K=8..15 & 24..31.  B-matrix 32x16: lanes 0-15 hold
// K=0..7 & 16..23 for column N=lane, lanes 16-31 hold K=8..15 & 24..31.
__global__ __launch_bounds__(256) void dfl_decode_wmma(
    const float* __restrict__ boxes, float* __restrict__ pb, int B, int N)
{
  const int tilesPerBatch = (N + 15) >> 4;
  const int totalWaves = B * tilesPerBatch;
  int gw = blockIdx.x * (blockDim.x >> 5) + (threadIdx.x >> 5);
  if (gw >= totalWaves) return;                 // wave-uniform exit (EXEC safe)

  int b  = gw / tilesPerBatch;
  int n0 = (gw - b * tilesPerBatch) << 4;

  int lane  = threadIdx.x & 31;
  int m     = lane & 15;                        // anchor-in-tile / B column
  int hi    = lane >> 4;                        // lane half
  int kbase = hi << 3;                          // bin base for A half
  int n = n0 + m; if (n > N - 1) n = N - 1;     // clamp (branchless select)

  const float* base = boxes + (size_t)b * 64 * N + n;

  // Normalized softmax probabilities per side; max and sum are reduced
  // across the two lane halves with shfl_xor(16) (ds_bpermute, wave32).
  float p[4][8];
  #pragma unroll
  for (int s = 0; s < 4; ++s) {
    float x[8];
    #pragma unroll
    for (int j = 0; j < 8; ++j)
      x[j] = base[(size_t)(s * 16 + kbase + j) * N];
    float mx = x[0];
    #pragma unroll
    for (int j = 1; j < 8; ++j) mx = fmaxf(mx, x[j]);
    mx = fmaxf(mx, __shfl_xor(mx, 16, 32));
    float sum = 0.0f;
    #pragma unroll
    for (int j = 0; j < 8; ++j) { p[s][j] = __expf(x[j] - mx); sum += p[s][j]; }
    sum += __shfl_xor(sum, 16, 32);
    float r = fast_rcp(sum);
    #pragma unroll
    for (int j = 0; j < 8; ++j) p[s][j] *= r;
  }

  // A fragments: A0 = sides 0,1 (K 0..15 / 16..31); A1 = sides 2,3.
  v16h a0, a1;
  #pragma unroll
  for (int t = 0; t < 8; ++t) {
    a0[t]     = (_Float16)p[0][t];
    a0[t + 8] = (_Float16)p[1][t];
    a1[t]     = (_Float16)p[2][t];
    a1[t + 8] = (_Float16)p[3][t];
  }

  // B fragment: col0 = proj (r = 0..15) for K<16, col1 = proj for K>=16.
  v16h bf;
  int klo = hi << 3;
  #pragma unroll
  for (int t = 0; t < 16; ++t) {
    int K = klo + t + (t & 8);
    int bin = K & 15, sh = K >> 4;
    float val = 0.0f;
    if      (m == 0 && sh == 0) val = (float)bin;
    else if (m == 1 && sh == 1) val = (float)bin;
    bf[t] = (_Float16)val;
  }

  v8f c = {};
  v8f d0 = __builtin_amdgcn_wmma_f32_16x16x32_f16(false, a0, false, bf,
                                                  (short)0, c, false, false);
  v8f d1 = __builtin_amdgcn_wmma_f32_16x16x32_f16(false, a1, false, bf,
                                                  (short)0, c, false, false);

  // D layout: lane holds column m, VGPR j = row j (+8 for hi lanes).
  // Columns 0/1 already hold E[side] — store directly from lanes 0,1,16,17.
  if (m < 2) {
    #pragma unroll
    for (int j = 0; j < 8; ++j) {
      int anchor = n0 + j + (hi << 3);
      if (anchor < N) {
        float* o = pb + ((size_t)b * N + anchor) * 4;
        o[m]     = d0[j];                        // sides 0/1
        o[m + 2] = d1[j];                        // sides 2/3
      }
    }
  }
}

// ------------------------------------------------- K2: CIoU + top-k scatter
// One block per (b,m). 256 threads stream N anchors; each keeps a sorted
// register top-10, then 10 rounds of block argmax extract the block top-10;
// positive values scatter via int-punned atomicMax (valid for floats > 0).
__global__ __launch_bounds__(256) void ciou_topk(
    const float* __restrict__ pb, const float* __restrict__ targets,
    float* __restrict__ iou_t, int B, int N, int M)
{
  __shared__ float sval[256];
  __shared__ int   sanc[256];
  __shared__ int   stid[256];

  int b = blockIdx.x / M;
  int m = blockIdx.x - b * M;
  const float* t = targets + ((size_t)b * M + m) * 4;
  float x21 = t[0], y21 = t[1], x22 = t[2], y22 = t[3];
  float w2 = x22 - x21, h2 = y22 - y21;
  float area2 = w2 * h2;
  float at2 = atanf(fast_div(w2, h2 + EPS_));
  const float four_over_pi2 = 0.405284734569f;   // 4/pi^2

  const float4* pb4 = (const float4*)pb;
  size_t bN = (size_t)b * N;
  int tid = threadIdx.x;

  float tv[TOPK]; int ti[TOPK];
  #pragma unroll
  for (int j = 0; j < TOPK; ++j) { tv[j] = NEG_INF_; ti[j] = 0; }

  for (int nn = tid; nn < N; nn += 256) {
    __builtin_prefetch(&pb4[bN + nn + 256], 0, 0);   // global_prefetch_b8
    float4 p = pb4[bN + nn];
    float x11 = p.x, y11 = p.y, x12 = p.z, y12 = p.w;
    float w1 = x12 - x11, h1 = y12 - y11;
    float iw = fmaxf(fminf(x12, x22) - fmaxf(x11, x21), 0.0f);
    float ih = fmaxf(fminf(y12, y22) - fmaxf(y11, y21), 0.0f);
    float inter = iw * ih;
    float uni = w1 * h1 + area2 - inter + EPS_;
    float iou = inter * fast_rcp(uni);
    float cw = fmaxf(x12, x22) - fminf(x11, x21);
    float ch = fmaxf(y12, y22) - fminf(y11, y21);
    float c2 = cw * cw + ch * ch + EPS_;
    float dx = x21 + x22 - x11 - x12;
    float dy = y21 + y22 - y11 - y12;
    float rho2 = (dx * dx + dy * dy) * 0.25f;
    float da = at2 - atanf(fast_div(w1, h1 + EPS_));
    float v = four_over_pi2 * da * da;
    float alpha = v * fast_rcp(v - iou + 1.0f + EPS_);
    float cio = iou - rho2 * fast_rcp(c2) - alpha * v;

    if (cio > tv[TOPK - 1]) {                    // bubble into sorted list
      float cv = cio; int ci = nn;
      #pragma unroll
      for (int j = 0; j < TOPK; ++j) {
        if (cv > tv[j]) {
          float tf = tv[j]; tv[j] = cv; cv = tf;
          int   tI = ti[j]; ti[j] = ci; ci = tI;
        }
      }
    }
  }

  for (int it = 0; it < TOPK; ++it) {
    sval[tid] = tv[0]; sanc[tid] = ti[0]; stid[tid] = tid;
    __syncthreads();
    for (int off = 128; off > 0; off >>= 1) {
      if (tid < off && sval[tid + off] > sval[tid]) {
        sval[tid] = sval[tid + off];
        sanc[tid] = sanc[tid + off];
        stid[tid] = stid[tid + off];
      }
      __syncthreads();
    }
    float wv = sval[0]; int wa = sanc[0]; int wt = stid[0];
    __syncthreads();
    if (tid == wt) {                             // winner pops its head
      #pragma unroll
      for (int j = 0; j < TOPK - 1; ++j) { tv[j] = tv[j + 1]; ti[j] = ti[j + 1]; }
      tv[TOPK - 1] = NEG_INF_;
    }
    if (tid == 0 && wv > 0.0f)
      atomicMax((int*)&iou_t[bN + wa], __float_as_int(wv));
  }
}

// ---------------------------------------------------- K3: per-batch reduce
__global__ __launch_bounds__(256) void batch_reduce(
    const float* __restrict__ iou_t, const float* __restrict__ scores,
    float* __restrict__ box_per, float* __restrict__ obj_per, int N)
{
  __shared__ float s0[256], s1[256], s2[256];
  int b = blockIdx.x, tid = threadIdx.x;
  size_t bN = (size_t)b * N;
  float np = 0.0f, sb = 0.0f, so = 0.0f;
  for (int n = tid; n < N; n += 256) {
    float t = iou_t[bN + n];
    float s = scores[bN + n];
    if (t > 0.0f) { np += 1.0f; sb += 1.0f - t; }
    so += fmaxf(s, 0.0f) - s * t + log1pf(expf(-fabsf(s)));
  }
  s0[tid] = np; s1[tid] = sb; s2[tid] = so;
  __syncthreads();
  for (int off = 128; off > 0; off >>= 1) {
    if (tid < off) {
      s0[tid] += s0[tid + off];
      s1[tid] += s1[tid + off];
      s2[tid] += s2[tid + off];
    }
    __syncthreads();
  }
  if (tid == 0) {
    box_per[b] = (s0[0] > 0.0f) ? s1[0] / s0[0] : 0.0f;
    obj_per[b] = s2[0] / (float)N;
  }
}

// -------------------------------------------------------- K4: finalize
__global__ void finalize(const float* __restrict__ box_per,
                         const float* __restrict__ obj_per,
                         float* __restrict__ out, int B)
{
  if (blockIdx.x == 0 && threadIdx.x == 0) {
    float bl = 0.0f, ol = 0.0f;
    for (int b = 0; b < B; ++b) { bl += box_per[b]; ol += obj_per[b]; }
    out[0] = (7.5f * bl + 1.0f * ol) / (float)B;   // total
    out[1] = bl;                                   // box_loss
    out[2] = ol;                                   // obj_loss
  }
}

// ---------------------------------------------------------------------------
extern "C" void kernel_launch(void* const* d_in, const int* in_sizes, int n_in,
                              void* d_out, int out_size, void* d_ws, size_t ws_size,
                              hipStream_t stream)
{
  const float* boxes   = (const float*)d_in[0];  // [B, 64, N]
  const float* scores  = (const float*)d_in[1];  // [B, N]
  const float* targets = (const float*)d_in[2];  // [B, M, 4]

  const int B = 8;
  const int N = in_sizes[1] / B;
  const int M = in_sizes[2] / (B * 4);

  float* pb      = (float*)d_ws;                  // [B,N,4]
  float* iou_t   = pb + (size_t)B * N * 4;        // [B,N]
  float* box_per = iou_t + (size_t)B * N;         // [B]
  float* obj_per = box_per + B;                   // [B]

  // zero iou_target (ws is poisoned, not re-zeroed between replays)
  {
    int n = B * N;
    zero_f32<<<(n + 255) / 256, 256, 0, stream>>>(iou_t, n);
  }
  // decode: one wave per 16-anchor tile, 8 waves per block
  {
    int tilesPerBatch = (N + 15) >> 4;
    int waves = B * tilesPerBatch;
    int blocks = (waves + 7) / 8;
    dfl_decode_wmma<<<blocks, 256, 0, stream>>>(boxes, pb, B, N);
  }
  // CIoU + top-k scatter: one block per (b, m)
  ciou_topk<<<B * M, 256, 0, stream>>>(pb, targets, iou_t, B, N, M);
  // per-batch reductions
  batch_reduce<<<B, 256, 0, stream>>>(iou_t, scores, box_per, obj_per, N);
  // deterministic scalar combine
  finalize<<<1, 32, 0, stream>>>(box_per, obj_per, (float*)d_out, B);
}
